// _BidirectionalSAGELayer_72954314490489
// MI455X (gfx1250) — matrix operand order repaired and checked by reading, not compiled, not run on hardware
//
#include <hip/hip_runtime.h>
#include <hip/hip_bf16.h>

typedef __bf16 bf16_t;
typedef bf16_t v16bf __attribute__((ext_vector_type(16)));
typedef float  v8f   __attribute__((ext_vector_type(8)));

#define N_NODES_   20000
#define IN_DIM_    256
#define HALF_DIM_  128
#define OUT_DIM_   256
#define N_EDGES_   320000
#define BN_EPS_    1e-5f

// ---- workspace layout (float element offsets) ----
#define AGG_         (N_NODES_*IN_DIM_)          // 5,120,000 floats per aggr buffer
#define CNT_PAD_     20480
#define OFF_AF_      0
#define OFF_AB_      (AGG_)
#define OFF_CF_      (2*AGG_)
#define OFF_CB_      (2*AGG_ + CNT_PAD_)
#define OFF_SUM_     (2*AGG_ + 2*CNT_PAD_)
#define OFF_SQ_      (OFF_SUM_ + 256)
#define ZERO_FLOATS_ (OFF_SQ_ + 256)             // region zeroed each call
#define OFF_SCALE_   (ZERO_FLOATS_)
#define OFF_SHIFT_   (OFF_SCALE_ + 256)
#define FLOAT_TOTAL_ (OFF_SHIFT_ + 256)
#define WB_BYTE_OFF_ ((size_t)FLOAT_TOTAL_ * 4)  // bf16 weights start here (32B aligned)
#define WMAT_        (HALF_DIM_*IN_DIM_)         // 32768 elements per weight matrix

// ---------------- zero accumulators ----------------
__global__ void k_zero(float4* p, int n4) {
    int i = blockIdx.x * blockDim.x + threadIdx.x;
    if (i < n4) p[i] = float4{0.f, 0.f, 0.f, 0.f};
}

// ---------------- f32 -> bf16 weight conversion ----------------
__global__ void k_cvtw(const float* wlf, const float* wrf,
                       const float* wlb, const float* wrb, bf16_t* out) {
    int i = blockIdx.x * blockDim.x + threadIdx.x;
    if (i < 4 * WMAT_) {
        int m = i / WMAT_, r = i % WMAT_;
        const float* src = (m == 0) ? wlf : (m == 1) ? wrf : (m == 2) ? wlb : wrb;
        out[i] = (bf16_t)src[r];
    }
}

// ---------------- edge scatter (both directions) ----------------
// one wave32 per edge; lane owns 8 contiguous channels
__global__ __launch_bounds__(256) void k_edges(const long long* ei, const float* x,
                                               float* aggr_f, float* aggr_b,
                                               float* cnt_f, float* cnt_b) {
    int warp = (blockIdx.x * blockDim.x + threadIdx.x) >> 5;
    int lane = threadIdx.x & 31;
    if (warp >= N_EDGES_) return;
    int src = (int)ei[warp];
    int dst = (int)ei[N_EDGES_ + warp];
    const float4* xs = (const float4*)(x + (size_t)src * IN_DIM_);
    const float4* xd = (const float4*)(x + (size_t)dst * IN_DIM_);
    float* af = aggr_f + (size_t)dst * IN_DIM_;
    float* ab = aggr_b + (size_t)src * IN_DIM_;
    int c4 = lane * 2;
    float4 v0 = xs[c4], v1 = xs[c4 + 1];
    float4 w0 = xd[c4], w1 = xd[c4 + 1];
    int cb = lane * 8;
    atomicAdd(af + cb + 0, v0.x); atomicAdd(af + cb + 1, v0.y);
    atomicAdd(af + cb + 2, v0.z); atomicAdd(af + cb + 3, v0.w);
    atomicAdd(af + cb + 4, v1.x); atomicAdd(af + cb + 5, v1.y);
    atomicAdd(af + cb + 6, v1.z); atomicAdd(af + cb + 7, v1.w);
    atomicAdd(ab + cb + 0, w0.x); atomicAdd(ab + cb + 1, w0.y);
    atomicAdd(ab + cb + 2, w0.z); atomicAdd(ab + cb + 3, w0.w);
    atomicAdd(ab + cb + 4, w1.x); atomicAdd(ab + cb + 5, w1.y);
    atomicAdd(ab + cb + 6, w1.z); atomicAdd(ab + cb + 7, w1.w);
    if (lane == 0) {
        atomicAdd(cnt_f + dst, 1.0f);
        atomicAdd(cnt_b + src, 1.0f);
    }
}

// ---------------- fused dual-SAGE GEMM (WMMA bf16) + bias + BN-stat accumulation ----
// block = 256 threads (8 waves); wave w: dir = w>>2 (fwd/bwd), owns 32 output cols.
// M tile = 16 nodes. K = 512 = [aggr/cnt (256) | x (256)].
__global__ __launch_bounds__(256) void k_gemm(const float* x,
                                              const float* aggr_f, const float* aggr_b,
                                              const float* cnt_f, const float* cnt_b,
                                              const bf16_t* wbf,
                                              const float* blf, const float* blb,
                                              float* chsum, float* chsumsq, float* out) {
    const int m0   = blockIdx.x * 16;
    const int tid  = threadIdx.x;
    const int w    = tid >> 5;
    const int lane = tid & 31;
    const int dir  = w >> 2;       // 0 = forward, 1 = backward
    const int wd   = w & 3;        // wave within direction
    const int half = lane >> 4;    // lane group 0: lanes 0-15, 1: lanes 16-31
    const int lc   = lane & 15;
    const int row  = m0 + lc;      // this lane's A-matrix row (node)

    const float*  aggr = dir ? aggr_b : aggr_f;
    const float*  cnt  = dir ? cnt_b  : cnt_f;
    const bf16_t* Wl   = wbf + (size_t)(dir ? 2 : 0) * WMAT_;
    const bf16_t* Wr   = wbf + (size_t)(dir ? 3 : 1) * WMAT_;
    const float*  bias = dir ? blb : blf;

    const float rc = 1.0f / fmaxf(cnt[row], 1.0f);
    const float* arow = aggr + (size_t)row * IN_DIM_;
    const float* xrow = x    + (size_t)row * IN_DIM_;

    const int o0 = wd * 32 + lc;   // B-fragment column (within direction, 0..127)
    const int o1 = o0 + 16;

    v8f acc0 = {0.f,0.f,0.f,0.f,0.f,0.f,0.f,0.f};
    v8f acc1 = {0.f,0.f,0.f,0.f,0.f,0.f,0.f,0.f};

    #pragma unroll
    for (int phase = 0; phase < 2; ++phase) {
        const float*  srow = phase ? xrow : arow;
        const bf16_t* Ws   = phase ? Wr : Wl;
        const float   sc   = phase ? 1.0f : rc;
        #pragma unroll
        for (int kt = 0; kt < 8; ++kt) {
            const int kb = kt * 32;
            // A fragment: lanes 0-15 -> K {kb+0..15}; lanes 16-31 -> K {kb+16..31}
            const float* s0 = srow + kb + half * 16;
            float4 f0 = *(const float4*)(s0);
            float4 f1 = *(const float4*)(s0 + 4);
            float4 f2 = *(const float4*)(s0 + 8);
            float4 f3 = *(const float4*)(s0 + 12);
            v16bf a;
            a[0]  = (bf16_t)(f0.x * sc); a[1]  = (bf16_t)(f0.y * sc);
            a[2]  = (bf16_t)(f0.z * sc); a[3]  = (bf16_t)(f0.w * sc);
            a[4]  = (bf16_t)(f1.x * sc); a[5]  = (bf16_t)(f1.y * sc);
            a[6]  = (bf16_t)(f1.z * sc); a[7]  = (bf16_t)(f1.w * sc);
            a[8]  = (bf16_t)(f2.x * sc); a[9]  = (bf16_t)(f2.y * sc);
            a[10] = (bf16_t)(f2.z * sc); a[11] = (bf16_t)(f2.w * sc);
            a[12] = (bf16_t)(f3.x * sc); a[13] = (bf16_t)(f3.y * sc);
            a[14] = (bf16_t)(f3.z * sc); a[15] = (bf16_t)(f3.w * sc);
            // B fragment: lane holds 16 consecutive K of column o (weights are [o][k] row-major)
            const int kk = kb + half * 16;
            v16bf b0 = *(const v16bf*)(Ws + (size_t)o0 * IN_DIM_ + kk);
            v16bf b1 = *(const v16bf*)(Ws + (size_t)o1 * IN_DIM_ + kk);
            acc0 = __builtin_amdgcn_wmma_f32_16x16x32_bf16(false, a, false, b0,
                                                           (short)0, acc0, false, false);
            acc1 = __builtin_amdgcn_wmma_f32_16x16x32_bf16(false, a, false, b1,
                                                           (short)0, acc1, false, false);
        }
    }

    // bias + write pre-BN h to d_out + per-channel sum / sumsq
    #pragma unroll
    for (int t = 0; t < 2; ++t) {
        const v8f   c      = t ? acc1 : acc0;
        const int   o_loc  = wd * 32 + t * 16 + lc;       // 0..127 within direction
        const int   o_glob = dir * HALF_DIM_ + o_loc;     // 0..255 output channel
        const float bv     = bias[o_loc];
        float* orow = out + (size_t)(m0 + half * 8) * OUT_DIM_ + o_glob;
        float s1 = 0.f, s2 = 0.f;
        #pragma unroll
        for (int i = 0; i < 8; ++i) {                     // C/D layout: VGPR i -> M = i + half*8
            float v = c[i] + bv;
            s1 += v;
            s2 += v * v;
            orow[(size_t)i * OUT_DIM_] = v;
        }
        s1 += __shfl_xor(s1, 16);                         // combine lane l with l^16 (same channel)
        s2 += __shfl_xor(s2, 16);
        if (half == 0) {
            atomicAdd(chsum + o_glob, s1);
            atomicAdd(chsumsq + o_glob, s2);
        }
    }
}

// ---------------- BN finalize: per-channel scale/shift ----------------
__global__ void k_bn(const float* chsum, const float* chsumsq,
                     const float* gamma, const float* beta,
                     float* scale, float* shift) {
    int o = threadIdx.x;
    const float inv_n = 1.0f / (float)N_NODES_;
    float m  = chsum[o] * inv_n;
    float v  = chsumsq[o] * inv_n - m * m;
    float sc = gamma[o] * rsqrtf(fmaxf(v, 0.0f) + BN_EPS_);
    scale[o] = sc;
    shift[o] = beta[o] - m * sc;
}

// ---------------- apply BN affine + ReLU in place on d_out ----------------
__global__ void k_apply(float* out, const float* scale, const float* shift) {
    int i = blockIdx.x * blockDim.x + threadIdx.x;
    if (i < N_NODES_ * OUT_DIM_ / 4) {
        int c = (i & 63) * 4;
        float4 v = ((float4*)out)[i];
        v.x = fmaxf(v.x * scale[c + 0] + shift[c + 0], 0.f);
        v.y = fmaxf(v.y * scale[c + 1] + shift[c + 1], 0.f);
        v.z = fmaxf(v.z * scale[c + 2] + shift[c + 2], 0.f);
        v.w = fmaxf(v.w * scale[c + 3] + shift[c + 3], 0.f);
        ((float4*)out)[i] = v;
    }
}

extern "C" void kernel_launch(void* const* d_in, const int* in_sizes, int n_in,
                              void* d_out, int out_size, void* d_ws, size_t ws_size,
                              hipStream_t stream) {
    const float*     x     = (const float*)d_in[0];
    const long long* ei    = (const long long*)d_in[1];   // int64 edge_index [2, E]
    const float*     wlf   = (const float*)d_in[2];
    const float*     blf   = (const float*)d_in[3];
    const float*     wrf   = (const float*)d_in[4];
    const float*     wlb   = (const float*)d_in[5];
    const float*     blb   = (const float*)d_in[6];
    const float*     wrb   = (const float*)d_in[7];
    const float*     gamma = (const float*)d_in[8];
    const float*     beta  = (const float*)d_in[9];
    float*           out   = (float*)d_out;

    float*  wsf = (float*)d_ws;
    bf16_t* wbf = (bf16_t*)((char*)d_ws + WB_BYTE_OFF_);

    float* aggr_f  = wsf + OFF_AF_;
    float* aggr_b  = wsf + OFF_AB_;
    float* cnt_f   = wsf + OFF_CF_;
    float* cnt_b   = wsf + OFF_CB_;
    float* chsum   = wsf + OFF_SUM_;
    float* chsumsq = wsf + OFF_SQ_;
    float* scale   = wsf + OFF_SCALE_;
    float* shift   = wsf + OFF_SHIFT_;

    // 1. zero accumulators
    {
        int n4 = ZERO_FLOATS_ / 4;
        k_zero<<<(n4 + 255) / 256, 256, 0, stream>>>((float4*)d_ws, n4);
    }
    // 2. weights -> bf16
    k_cvtw<<<(4 * WMAT_ + 255) / 256, 256, 0, stream>>>(wlf, wrf, wlb, wrb, wbf);
    // 3. edge aggregation (both directions), 8 edges (waves) per block
    k_edges<<<N_EDGES_ / 8, 256, 0, stream>>>(ei, x, aggr_f, aggr_b, cnt_f, cnt_b);
    // 4. fused WMMA GEMM + bias + BN stats -> pre-BN h in d_out
    k_gemm<<<N_NODES_ / 16, 256, 0, stream>>>(x, aggr_f, aggr_b, cnt_f, cnt_b,
                                              wbf, blf, blb, chsum, chsumsq, out);
    // 5. BN scale/shift
    k_bn<<<1, 256, 0, stream>>>(chsum, chsumsq, gamma, beta, scale, shift);
    // 6. affine + ReLU in place
    k_apply<<<(N_NODES_ * OUT_DIM_ / 4 + 255) / 256, 256, 0, stream>>>(out, scale, shift);

    (void)in_sizes; (void)n_in; (void)out_size; (void)ws_size;
}
